// Interaction_Att_82832739270704
// MI455X (gfx1250) — matrix operand order repaired and checked by reading, not compiled, not run on hardware
//
#include <hip/hip_runtime.h>

// ---------------------------------------------------------------------------
// Types for CDNA5 WMMA (wave32, v_wmma_f32_16x16x32_bf16)
// ---------------------------------------------------------------------------
typedef __bf16 bf16;
typedef __attribute__((ext_vector_type(16))) __bf16 v16bf;
typedef __attribute__((ext_vector_type(8)))  __bf16 v8bf;
typedef __attribute__((ext_vector_type(8)))  float  v8f;

#define WMMA_BF16(a, b, c) \
  __builtin_amdgcn_wmma_f32_16x16x32_bf16(false, (a), false, (b), (short)0, (c), false, false)

__device__ __forceinline__ bf16 f2bf(float f) {
  union { float f; unsigned u; } x; x.f = f;
  unsigned r = x.u + 0x7FFFu + ((x.u >> 16) & 1u);   // round-to-nearest-even
  unsigned short h = (unsigned short)(r >> 16);
  bf16 out;
  __builtin_memcpy(&out, &h, 2);
  return out;
}

union FragU { v16bf v; v8bf h[2]; };

// A-matrix 16x32 bf16 fragment (ISA 7.12.2): lane half=lane>>4, row=lane&15;
// per-lane K runs: [half*8, half*8+8) and [16+half*8, 24+half*8)
__device__ __forceinline__ v16bf load_a_frag(const bf16* __restrict__ base, int stride, int kk) {
  int lane = threadIdx.x & 31;
  int r = lane & 15, half = lane >> 4;
  const bf16* p = base + (size_t)r * stride + kk + half * 8;
  FragU f;
  f.h[0] = *reinterpret_cast<const v8bf*>(p);
  f.h[1] = *reinterpret_cast<const v8bf*>(p + 16);
  return f.v;
}

// B-matrix 32x16 bf16 fragment: lane col=lane&15; per-lane K runs [half*8, half*8+8) and [16+half*8, 24+half*8)
// `base` points at a [N,K]-major ("B-transposed") array: column n's K-vector is contiguous.
__device__ __forceinline__ v16bf load_b_frag(const bf16* __restrict__ base, int stride, int kk) {
  int lane = threadIdx.x & 31;
  int c = lane & 15, half = lane >> 4;
  const bf16* p = base + (size_t)c * stride + kk + half * 8;
  FragU f;
  f.h[0] = *reinterpret_cast<const v8bf*>(p);
  f.h[1] = *reinterpret_cast<const v8bf*>(p + 16);
  return f.v;
}

// ---------------------------------------------------------------------------
// Conversion kernels
// ---------------------------------------------------------------------------
// Convert activations [M,128] f32 -> bf16; also drop a copy into the concat
// buffer columns [128,256).
__global__ __launch_bounds__(256) void conv_act_kernel(const float* __restrict__ src,
                                                       bf16* __restrict__ dstA,
                                                       bf16* __restrict__ dstCat,
                                                       int M) {
  int i = blockIdx.x * blockDim.x + threadIdx.x;
  if (i >= M * 128) return;
  bf16 v = f2bf(src[i]);
  dstA[i] = v;
  int m = i >> 7, k = i & 127;
  dstCat[(size_t)m * 256 + 128 + k] = v;
}

// src f32 [K,N] row-major -> dst bf16 [N,K] row-major (weight transpose)
__global__ __launch_bounds__(256) void transpose_conv_kernel(const float* __restrict__ src,
                                                             bf16* __restrict__ dst,
                                                             int K, int N) {
  int i = blockIdx.x * blockDim.x + threadIdx.x;
  if (i >= K * N) return;
  int k = i / N, n = i % N;
  dst[(size_t)n * K + k] = f2bf(src[i]);
}

// ---------------------------------------------------------------------------
// Generic WMMA GEMM:  out = act( A[M,K] @ Bt[N,K]^T + bias[N] )
// Each wave computes a 16x64 strip (4 accumulators, A-fragment reuse x4).
// mode 0: bf16 store to out[m*ldo + coloff + n]
// mode 1: head-transposed bf16 store to out[((b*8 + n>>7)*128 + (n&127))*L + l]
// ---------------------------------------------------------------------------
__global__ __launch_bounds__(256) void gemm_bf16_kernel(const bf16* __restrict__ A,
                                                        const bf16* __restrict__ Bt,
                                                        const float* __restrict__ bias,
                                                        bf16* __restrict__ out,
                                                        int M, int N, int K,
                                                        int ldo, int coloff,
                                                        int relu, int mode, int L) {
  int wid = blockIdx.x * (blockDim.x >> 5) + (threadIdx.x >> 5);
  int mt = M >> 4, nt64 = N >> 6;
  if (wid >= mt * nt64) return;
  int tm = wid % mt, tn = wid / mt;
  int lane = threadIdx.x & 31;
  int col16 = lane & 15, half = lane >> 4;

  v8f zero = {0.f, 0.f, 0.f, 0.f, 0.f, 0.f, 0.f, 0.f};
  v8f acc[4];
#pragma unroll
  for (int t = 0; t < 4; ++t) acc[t] = zero;

  const bf16* Arow = A + (size_t)tm * 16 * K;
  for (int kk = 0; kk < K; kk += 32) {
    v16bf a = load_a_frag(Arow, K, kk);
#pragma unroll
    for (int t = 0; t < 4; ++t) {
      v16bf b = load_b_frag(Bt + (size_t)(tn * 64 + t * 16) * K, K, kk);
      acc[t] = WMMA_BF16(a, b, acc[t]);
    }
  }

#pragma unroll
  for (int t = 0; t < 4; ++t) {
    int n = tn * 64 + t * 16 + col16;
    float bv = bias ? bias[n] : 0.f;
#pragma unroll
    for (int r = 0; r < 8; ++r) {
      int m = tm * 16 + r + 8 * half;
      float v = acc[t][r] + bv;
      if (relu) v = fmaxf(v, 0.f);
      if (mode == 0) {
        out[(size_t)m * ldo + coloff + n] = f2bf(v);
      } else {
        int bidx = m / L, l = m % L;
        int hh = n >> 7, d = n & 127;
        out[(((size_t)bidx * 8 + hh) * 128 + d) * (size_t)L + l] = f2bf(v);
      }
    }
  }
}

// Final-stage WMMA GEMM with f32 relu output (writes to d_out).
__global__ __launch_bounds__(256) void gemm_f32_kernel(const bf16* __restrict__ A,
                                                       const bf16* __restrict__ Bt,
                                                       const float* __restrict__ bias,
                                                       float* __restrict__ out,
                                                       int M, int N, int K) {
  int wid = blockIdx.x * (blockDim.x >> 5) + (threadIdx.x >> 5);
  int mt = M >> 4, nt64 = N >> 6;
  if (wid >= mt * nt64) return;
  int tm = wid % mt, tn = wid / mt;
  int lane = threadIdx.x & 31;
  int col16 = lane & 15, half = lane >> 4;

  v8f zero = {0.f, 0.f, 0.f, 0.f, 0.f, 0.f, 0.f, 0.f};
  v8f acc[4];
#pragma unroll
  for (int t = 0; t < 4; ++t) acc[t] = zero;

  const bf16* Arow = A + (size_t)tm * 16 * K;
  for (int kk = 0; kk < K; kk += 32) {
    v16bf a = load_a_frag(Arow, K, kk);
#pragma unroll
    for (int t = 0; t < 4; ++t) {
      v16bf b = load_b_frag(Bt + (size_t)(tn * 64 + t * 16) * K, K, kk);
      acc[t] = WMMA_BF16(a, b, acc[t]);
    }
  }

#pragma unroll
  for (int t = 0; t < 4; ++t) {
    int n = tn * 64 + t * 16 + col16;
    float bv = bias[n];
#pragma unroll
    for (int r = 0; r < 8; ++r) {
      int m = tm * 16 + r + 8 * half;
      out[(size_t)m * N + n] = fmaxf(acc[t][r] + bv, 0.f);
    }
  }
}

// ---------------------------------------------------------------------------
// Attention kernel: one block per (b, h, 16-row ligand tile).
// Phase 1: energy = (Q @ K^T)/sqrt(128) * inter      (WMMA, regs)
// Phase 2: row softmax (shfl_xor + small LDS cross-wave reduce)
// Phase 3: att -> LDS (bf16) and att^T -> global workspace
// Phase 4: lig tile = att @ P2  (WMMA, A-fragments from LDS)
// ---------------------------------------------------------------------------
__global__ __launch_bounds__(256) void attn_kernel(const bf16* __restrict__ l1,
                                                   const bf16* __restrict__ p1,
                                                   const bf16* __restrict__ p2t,
                                                   const float* __restrict__ inter,
                                                   bf16* __restrict__ att_t,
                                                   bf16* __restrict__ lig3_pre) {
  __shared__ __align__(32) bf16 att_s[16 * 1024];  // 32 KB
  __shared__ float red[8][16];

  int blk = blockIdx.x;            // B * H * (LL/16) = 16*8*16
  int b = blk >> 7;
  int rem = blk & 127;
  int h = rem >> 4;
  int l0 = (rem & 15) << 4;

  int w = threadIdx.x >> 5;
  int lane = threadIdx.x & 31;
  int col16 = lane & 15, half = lane >> 4;

  const bf16* Qb = l1 + ((size_t)(b * 256 + l0)) * 1024 + h * 128;   // rows l, stride 1024
  const bf16* Kb = p1 + ((size_t)b * 1024) * 1024 + h * 128;         // rows p, stride 1024

  v8f zero = {0.f, 0.f, 0.f, 0.f, 0.f, 0.f, 0.f, 0.f};
  v8f acc[8];
#pragma unroll
  for (int t = 0; t < 8; ++t) acc[t] = zero;

  // energy: wave w covers p in [w*128, w*128+128)
  for (int kk = 0; kk < 128; kk += 32) {
    v16bf a = load_a_frag(Qb, 1024, kk);
#pragma unroll
    for (int t = 0; t < 8; ++t) {
      v16bf bb = load_b_frag(Kb + (size_t)(w * 128 + t * 16) * 1024, 1024, kk);
      acc[t] = WMMA_BF16(a, bb, acc[t]);
    }
  }

  const float scale = 0.088388347648318447f;  // 1/sqrt(128)
  float rmax[8];
#pragma unroll
  for (int r = 0; r < 8; ++r) rmax[r] = -3.0e38f;

#pragma unroll
  for (int t = 0; t < 8; ++t) {
    int p = w * 128 + t * 16 + col16;
#pragma unroll
    for (int r = 0; r < 8; ++r) {
      int l = l0 + r + 8 * half;
      float v = acc[t][r] * scale * inter[((size_t)b * 256 + l) * 1024 + p];
      acc[t][r] = v;
      rmax[r] = fmaxf(rmax[r], v);
    }
  }
  // reduce max across 16 lanes sharing a row
#pragma unroll
  for (int r = 0; r < 8; ++r)
    for (int off = 1; off < 16; off <<= 1)
      rmax[r] = fmaxf(rmax[r], __shfl_xor(rmax[r], off, 32));
  if (col16 == 0) {
#pragma unroll
    for (int r = 0; r < 8; ++r) red[w][half * 8 + r] = rmax[r];
  }
  __syncthreads();
#pragma unroll
  for (int r = 0; r < 8; ++r) {
    float m = red[0][half * 8 + r];
#pragma unroll
    for (int ww = 1; ww < 8; ++ww) m = fmaxf(m, red[ww][half * 8 + r]);
    rmax[r] = m;
  }
  __syncthreads();

  float rsum[8];
#pragma unroll
  for (int r = 0; r < 8; ++r) rsum[r] = 0.f;
#pragma unroll
  for (int t = 0; t < 8; ++t) {
#pragma unroll
    for (int r = 0; r < 8; ++r) {
      float e = __expf(acc[t][r] - rmax[r]);
      acc[t][r] = e;
      rsum[r] += e;
    }
  }
#pragma unroll
  for (int r = 0; r < 8; ++r)
    for (int off = 1; off < 16; off <<= 1)
      rsum[r] += __shfl_xor(rsum[r], off, 32);
  if (col16 == 0) {
#pragma unroll
    for (int r = 0; r < 8; ++r) red[w][half * 8 + r] = rsum[r];
  }
  __syncthreads();
#pragma unroll
  for (int r = 0; r < 8; ++r) {
    float s = 0.f;
#pragma unroll
    for (int ww = 0; ww < 8; ++ww) s += red[ww][half * 8 + r];
    rsum[r] = 1.0f / s;
  }

  // write att (LDS) and att^T (global)
#pragma unroll
  for (int t = 0; t < 8; ++t) {
    int p = w * 128 + t * 16 + col16;
#pragma unroll
    for (int r = 0; r < 8; ++r) {
      int row = r + 8 * half;
      bf16 av = f2bf(acc[t][r] * rsum[r]);
      att_s[(size_t)row * 1024 + p] = av;
      att_t[(((size_t)b * 8 + h) * 1024 + p) * 256 + (l0 + row)] = av;
    }
  }
  __syncthreads();

  // lig tile: att[16,1024] @ P2[1024,128]; wave w does d-cols [w*16, w*16+16)
  int d0 = w * 16;
  const bf16* Bb = p2t + (((size_t)b * 8 + h) * 128 + d0) * 1024;  // rows d, stride 1024 (contig p)
  v8f oc = zero;
  for (int kk = 0; kk < 1024; kk += 32) {
    v16bf a = load_a_frag(att_s, 1024, kk);
    v16bf bb = load_b_frag(Bb, 1024, kk);
    oc = WMMA_BF16(a, bb, oc);
  }
#pragma unroll
  for (int r = 0; r < 8; ++r) {
    int row = r + 8 * half;
    lig3_pre[((size_t)(b * 256 + l0 + row)) * 1024 + h * 128 + d0 + col16] = f2bf(oc[r]);
  }
}

// ---------------------------------------------------------------------------
// Protein branch: prot3_pre[b,p,h*128+d] = sum_l att[l,p] * l2[l,d]
// A = att_t rows (contiguous K=l), B = l2t rows (contiguous K=l).
// One block per (b, h, 16-row protein tile); wave w does 16 d-cols; K=256.
// ---------------------------------------------------------------------------
__global__ __launch_bounds__(256) void prot_attn_kernel(const bf16* __restrict__ att_t,
                                                        const bf16* __restrict__ l2t,
                                                        bf16* __restrict__ prot3_pre) {
  int blk = blockIdx.x;            // B * H * (LP/16) = 16*8*64
  int b = blk >> 9;
  int rem = blk & 511;
  int h = rem >> 6;
  int p0 = (rem & 63) << 4;

  int lane = threadIdx.x & 31;
  int w = threadIdx.x >> 5;
  int col16 = lane & 15, half = lane >> 4;
  int d0 = w * 16;

  const bf16* Ab = att_t + (((size_t)b * 8 + h) * 1024 + p0) * 256;  // rows p, stride 256
  const bf16* Bb = l2t + (((size_t)b * 8 + h) * 128 + d0) * 256;     // rows d, stride 256

  v8f oc = {0.f, 0.f, 0.f, 0.f, 0.f, 0.f, 0.f, 0.f};
  for (int kk = 0; kk < 256; kk += 32) {
    v16bf a = load_a_frag(Ab, 256, kk);
    v16bf bb = load_b_frag(Bb, 256, kk);
    oc = WMMA_BF16(a, bb, oc);
  }
#pragma unroll
  for (int r = 0; r < 8; ++r) {
    int row = r + 8 * half;
    prot3_pre[((size_t)(b * 1024 + p0 + row)) * 1024 + h * 128 + d0 + col16] = f2bf(oc[r]);
  }
}

// ---------------------------------------------------------------------------
// Host launch
// ---------------------------------------------------------------------------
extern "C" void kernel_launch(void* const* d_in, const int* in_sizes, int n_in,
                              void* d_out, int out_size, void* d_ws, size_t ws_size,
                              hipStream_t stream) {
  (void)in_sizes; (void)n_in; (void)out_size; (void)ws_size;
  const float* ligand = (const float*)d_in[0];
  const float* prot   = (const float*)d_in[1];
  const float* inter  = (const float*)d_in[2];
  const float* Wl1 = (const float*)d_in[3];  const float* bl1 = (const float*)d_in[4];
  const float* Wl2 = (const float*)d_in[5];  const float* bl2 = (const float*)d_in[6];
  const float* Wp1 = (const float*)d_in[7];  const float* bp1 = (const float*)d_in[8];
  const float* Wp2 = (const float*)d_in[9];  const float* bp2 = (const float*)d_in[10];
  const float* W11 = (const float*)d_in[11]; const float* b11 = (const float*)d_in[12];
  const float* W12 = (const float*)d_in[13]; const float* b12 = (const float*)d_in[14];
  const float* W21 = (const float*)d_in[15]; const float* b21 = (const float*)d_in[16];
  const float* W22 = (const float*)d_in[17]; const float* b22 = (const float*)d_in[18];
  float* out = (float*)d_out;

  // bump allocator over workspace (bf16 elements, 256B-aligned chunks)
  bf16* ws = reinterpret_cast<bf16*>(d_ws);
  size_t off = 0;
  auto alloc = [&](size_t n) { bf16* p = ws + off; off += (n + 127) & ~(size_t)127; return p; };

  bf16* lig_bf    = alloc((size_t)4096 * 128);
  bf16* prot_bf   = alloc((size_t)16384 * 128);
  bf16* cat_l     = alloc((size_t)4096 * 256);
  bf16* cat_p     = alloc((size_t)16384 * 256);
  bf16* Wl1t      = alloc((size_t)1024 * 128);
  bf16* Wl2t      = alloc((size_t)1024 * 128);
  bf16* Wp1t      = alloc((size_t)1024 * 128);
  bf16* Wp2t      = alloc((size_t)1024 * 128);
  bf16* W11t      = alloc((size_t)128 * 1024);
  bf16* W21t      = alloc((size_t)128 * 1024);
  bf16* W12t      = alloc((size_t)128 * 256);
  bf16* W22t      = alloc((size_t)128 * 256);
  bf16* l1buf     = alloc((size_t)4096 * 1024);
  bf16* p1buf     = alloc((size_t)16384 * 1024);
  bf16* l2t       = alloc((size_t)16 * 8 * 128 * 256);
  bf16* p2t       = alloc((size_t)16 * 8 * 128 * 1024);
  bf16* att_t     = alloc((size_t)16 * 8 * 1024 * 256);
  bf16* lig3_pre  = alloc((size_t)4096 * 1024);
  bf16* prot3_pre = alloc((size_t)16384 * 1024);

  // --- conversions ---
  conv_act_kernel<<<2048, 256, 0, stream>>>(ligand, lig_bf, cat_l, 4096);
  conv_act_kernel<<<8192, 256, 0, stream>>>(prot, prot_bf, cat_p, 16384);
  transpose_conv_kernel<<<512, 256, 0, stream>>>(Wl1, Wl1t, 128, 1024);
  transpose_conv_kernel<<<512, 256, 0, stream>>>(Wl2, Wl2t, 128, 1024);
  transpose_conv_kernel<<<512, 256, 0, stream>>>(Wp1, Wp1t, 128, 1024);
  transpose_conv_kernel<<<512, 256, 0, stream>>>(Wp2, Wp2t, 128, 1024);
  transpose_conv_kernel<<<512, 256, 0, stream>>>(W11, W11t, 1024, 128);
  transpose_conv_kernel<<<512, 256, 0, stream>>>(W21, W21t, 1024, 128);
  transpose_conv_kernel<<<128, 256, 0, stream>>>(W12, W12t, 256, 128);
  transpose_conv_kernel<<<128, 256, 0, stream>>>(W22, W22t, 256, 128);

  // --- projections (relu) ---
  gemm_bf16_kernel<<<512, 256, 0, stream>>>(lig_bf, Wl1t, bl1, l1buf, 4096, 1024, 128, 1024, 0, 1, 0, 0);
  gemm_bf16_kernel<<<512, 256, 0, stream>>>(lig_bf, Wl2t, bl2, l2t,   4096, 1024, 128, 0,    0, 1, 1, 256);
  gemm_bf16_kernel<<<2048, 256, 0, stream>>>(prot_bf, Wp1t, bp1, p1buf, 16384, 1024, 128, 1024, 0, 1, 0, 0);
  gemm_bf16_kernel<<<2048, 256, 0, stream>>>(prot_bf, Wp2t, bp2, p2t,   16384, 1024, 128, 0,    0, 1, 1, 1024);

  // --- attention ---
  attn_kernel<<<2048, 256, 0, stream>>>(l1buf, p1buf, p2t, inter, att_t, lig3_pre);
  prot_attn_kernel<<<8192, 256, 0, stream>>>(att_t, l2t, prot3_pre);

  // --- output MLP stage 1 (bias, no relu) into concat buffers ---
  gemm_bf16_kernel<<<64, 256, 0, stream>>>(lig3_pre, W11t, b11, cat_l, 4096, 128, 1024, 256, 0, 0, 0, 0);
  gemm_bf16_kernel<<<256, 256, 0, stream>>>(prot3_pre, W21t, b21, cat_p, 16384, 128, 1024, 256, 0, 0, 0, 0);

  // --- output MLP stage 2 (relu, f32 out) ---
  gemm_f32_kernel<<<64, 256, 0, stream>>>(cat_l, W12t, b12, out, 4096, 128, 256);
  gemm_f32_kernel<<<256, 256, 0, stream>>>(cat_p, W22t, b22, out + (size_t)4096 * 128, 16384, 128, 256);
}
